// MILCellModelmerge_3444563771661
// MI455X (gfx1250) — matrix operand order repaired and checked by reading, not compiled, not run on hardware
//
#include <hip/hip_runtime.h>
#include <math.h>

// ---------------------------------------------------------------------------
// Model dims (compile-time)
// ---------------------------------------------------------------------------
constexpr int kB = 4, kN = 1024, kE = 512, kD = 256, kH = 2, kC = 128;
constexpr int kFF = 1024, kNQ = 4, kNCLS = 24;
constexpr float kScale = 0.08838834764831845f;   // 1/sqrt(128)

// ---------------------------------------------------------------------------
// WMMA types (probe-verified for gfx1250)
// ---------------------------------------------------------------------------
typedef _Float16 v16h __attribute__((ext_vector_type(16)));
typedef _Float16 v8h  __attribute__((ext_vector_type(8)));
typedef float    v8f  __attribute__((ext_vector_type(8)));

__device__ __forceinline__ v8h cvt8(float4 a, float4 b) {
  v8h r;
  r[0] = (_Float16)a.x; r[1] = (_Float16)a.y;
  r[2] = (_Float16)a.z; r[3] = (_Float16)a.w;
  r[4] = (_Float16)b.x; r[5] = (_Float16)b.y;
  r[6] = (_Float16)b.z; r[7] = (_Float16)b.w;
  return r;
}

// ---------------------------------------------------------------------------
// Block-wide reductions (256 threads)
// ---------------------------------------------------------------------------
__device__ __forceinline__ float block_reduce_sum(float v, float* red) {
  int tid = threadIdx.x;
  red[tid] = v; __syncthreads();
  for (int off = 128; off > 0; off >>= 1) {
    if (tid < off) red[tid] += red[tid + off];
    __syncthreads();
  }
  float r = red[0]; __syncthreads();
  return r;
}

__device__ __forceinline__ float block_reduce_max(float v, float* red) {
  int tid = threadIdx.x;
  red[tid] = v; __syncthreads();
  for (int off = 128; off > 0; off >>= 1) {
    if (tid < off) red[tid] = fmaxf(red[tid], red[tid + off]);
    __syncthreads();
  }
  float r = red[0]; __syncthreads();
  return r;
}

// ---------------------------------------------------------------------------
// Generic WMMA GEMM:
//   D[M,Nn] = act( alpha * A[M,K] @ op(B) + bias[Nn] )
//   op(B): btrans=1 -> B is [Nn,K] (row-major, computes A@B^T, "weight" form)
//          btrans=0 -> B is [K,Nn] (row-major, computes A@B,   "V" form)
//   Two-level batch via blockIdx.z: offset = (z/zInner)*sOut + (z%zInner)*sIn
//   Block = 256 threads = 8 wave32 waves; tile 128(M) x 64(N) x 32(K);
//   each wave owns a 32x32 micro-tile = 2x2 v_wmma_f32_16x16x32_f16.
//   Staging: branchless; vectorized b128 fast path for interior tiles;
//   clamp+select (no exec-mask serialization) on edge tiles.
// ---------------------------------------------------------------------------
#define ASTR 48   // LDS row stride in halves (96B: keeps 16B/32B chunks aligned)
#define BSTR 48

__global__ __launch_bounds__(256)
void gemm_f16_kernel(const float* __restrict__ A, const float* __restrict__ Bm,
                     const float* __restrict__ bias, float* __restrict__ Dst,
                     int M, int Nn, int K, int lda, int ldb, int ldc,
                     float alpha, int btrans, int relu, int zInner,
                     long aO, long aI, long bO, long bI, long dO, long dI) {
  __shared__ __align__(32) _Float16 As[128 * ASTR];
  __shared__ __align__(32) _Float16 Bs[64 * BSTR];

  const int tid  = threadIdx.x;
  const int lane = tid & 31;
  const int wid  = tid >> 5;
  const int wm   = wid & 3;       // 0..3  (row group of 32)
  const int wn   = wid >> 2;      // 0..1  (col group of 32)
  const int lrow = lane & 15;
  const int lhi  = lane >> 4;
  const int ka   = lhi ? 16 : 0;  // A frag: halves [ka..ka+15]
  const int kb   = lhi ? 16 : 0;  // B frag: halves [kb..kb+15]

  const int z  = blockIdx.z;
  const int zo = z / zInner;
  const int zi = z - zo * zInner;
  const float* Ab = A + (long)zo * aO + (long)zi * aI;
  const float* Bb = Bm + (long)zo * bO + (long)zi * bI;
  float*       Db = Dst + (long)zo * dO + (long)zi * dI;

  const int mBase = blockIdx.y * 128;
  const int nBase = blockIdx.x * 64;
  const bool fullM = (mBase + 128 <= M);
  const bool fullN = (nBase + 64 <= Nn);

  v8f acc[2][2];
  acc[0][0] = (v8f){}; acc[0][1] = (v8f){};
  acc[1][0] = (v8f){}; acc[1][1] = (v8f){};

  for (int k0 = 0; k0 < K; k0 += 32) {
    const bool fullK = (k0 + 32 <= K);
    __syncthreads();

    // ---- stage A tile 128x32 (fp32 -> f16) ----
    if (fullM && fullK) {
      // thread -> (row, 16-elem chunk); 4x b128 loads, 2x b128 LDS stores
      int r = tid >> 1, cb = (tid & 1) * 16;
      const float4* ap4 = (const float4*)(Ab + (long)(mBase + r) * lda + k0 + cb);
      float4 f0 = ap4[0], f1 = ap4[1], f2 = ap4[2], f3 = ap4[3];
      v8h* dst = (v8h*)&As[r * ASTR + cb];
      dst[0] = cvt8(f0, f1);
      dst[1] = cvt8(f2, f3);
    } else {
      for (int e = tid; e < 128 * 32; e += 256) {
        int r = e >> 5, c = e & 31;
        int gm = mBase + r, gk = k0 + c;
        int cm = gm < M ? gm : M - 1;     // clamped address: always valid,
        int ck = gk < K ? gk : K - 1;     // select-to-zero is branchless
        float v = Ab[(long)cm * lda + ck];
        if (gm >= M || gk >= K) v = 0.f;
        As[r * ASTR + c] = (_Float16)v;
      }
    }

    // ---- stage B tile as Bs[n][k] ----
    if (btrans) {
      if (fullN && fullK) {
        if (tid < 128) {
          int r = tid >> 1, cb = (tid & 1) * 16;
          const float4* bp4 = (const float4*)(Bb + (long)(nBase + r) * ldb + k0 + cb);
          float4 f0 = bp4[0], f1 = bp4[1], f2 = bp4[2], f3 = bp4[3];
          v8h* dst = (v8h*)&Bs[r * BSTR + cb];
          dst[0] = cvt8(f0, f1);
          dst[1] = cvt8(f2, f3);
        }
      } else {
        for (int e = tid; e < 64 * 32; e += 256) {
          int r = e >> 5, c = e & 31;
          int gn = nBase + r, gk = k0 + c;
          int cn = gn < Nn ? gn : Nn - 1;
          int ck = gk < K ? gk : K - 1;
          float v = Bb[(long)cn * ldb + ck];
          if (gn >= Nn || gk >= K) v = 0.f;
          Bs[r * BSTR + c] = (_Float16)v;
        }
      }
    } else {
      // B is [K, Nn] row-major: coalesce along n
      if (fullN && fullK) {
        for (int e = tid; e < 512; e += 256) {   // 32 k-rows x 16 float4 chunks
          int kk = e >> 4, ng = e & 15;
          float4 f = *(const float4*)(Bb + (long)(k0 + kk) * ldb + nBase + ng * 4);
          int nb = ng * 4;
          Bs[(nb + 0) * BSTR + kk] = (_Float16)f.x;
          Bs[(nb + 1) * BSTR + kk] = (_Float16)f.y;
          Bs[(nb + 2) * BSTR + kk] = (_Float16)f.z;
          Bs[(nb + 3) * BSTR + kk] = (_Float16)f.w;
        }
      } else {
        for (int e = tid; e < 64 * 32; e += 256) {
          int kk = e >> 6, nn = e & 63;
          int gn = nBase + nn, gk = k0 + kk;
          int cn = gn < Nn ? gn : Nn - 1;
          int ck = gk < K ? gk : K - 1;
          float v = Bb[(long)ck * ldb + cn];
          if (gn >= Nn || gk >= K) v = 0.f;
          Bs[nn * BSTR + kk] = (_Float16)v;
        }
      }
    }

    // ---- prefetch next K tile (global_prefetch_b8; speculative, safe) ----
    if (k0 + 32 < K) {
      if (tid < 128) {
        int pr = mBase + tid; if (pr > M - 1) pr = M - 1;
        __builtin_prefetch(Ab + (long)pr * lda + (k0 + 32), 0, 3);
      } else if (btrans) {
        int t2 = tid - 128;
        if (t2 < 64) {
          int pr = nBase + t2; if (pr > Nn - 1) pr = Nn - 1;
          __builtin_prefetch(Bb + (long)pr * ldb + (k0 + 32), 0, 3);
        }
      } else {
        int t2 = tid - 128;
        if (t2 < 32) {
          int pk = k0 + 32 + t2; if (pk > K - 1) pk = K - 1;
          __builtin_prefetch(Bb + (long)pk * ldb + nBase, 0, 3);
        }
      }
    }
    __syncthreads();

    // ---- 2x2 WMMA micro-tile ----
    union { v16h v; v8h h[2]; } af;
#pragma unroll
    for (int tm = 0; tm < 2; ++tm) {
      const _Float16* ap = &As[(wm * 32 + tm * 16 + lrow) * ASTR + ka];
      af.v = *(const v16h*)ap;
#pragma unroll
      for (int tn = 0; tn < 2; ++tn) {
        v16h bf = *(const v16h*)&Bs[(wn * 32 + tn * 16 + lrow) * BSTR + kb];
        acc[tm][tn] = __builtin_amdgcn_wmma_f32_16x16x32_f16(
            false, af.v, false, bf, (short)0, acc[tm][tn], false, false);
      }
    }
  }

  // epilogue: C/D layout VGPR r -> (M = r + 8*(lane>>4), N = lane&15)
  if (fullM && fullN) {
#pragma unroll
    for (int tm = 0; tm < 2; ++tm) {
#pragma unroll
      for (int tn = 0; tn < 2; ++tn) {
        int col = nBase + wn * 32 + tn * 16 + lrow;
        int rb  = mBase + wm * 32 + tm * 16 + lhi * 8;
        float bv = bias ? bias[col] : 0.f;
#pragma unroll
        for (int r = 0; r < 8; ++r) {
          float v = acc[tm][tn][r] * alpha + bv;
          if (relu) v = fmaxf(v, 0.f);
          Db[(long)(rb + r) * ldc + col] = v;
        }
      }
    }
  } else {
#pragma unroll
    for (int tm = 0; tm < 2; ++tm) {
#pragma unroll
      for (int tn = 0; tn < 2; ++tn) {
        int col = nBase + wn * 32 + tn * 16 + lrow;
        int rb  = mBase + wm * 32 + tm * 16 + lhi * 8;
        float bv = (bias && col < Nn) ? bias[col] : 0.f;
#pragma unroll
        for (int r = 0; r < 8; ++r) {
          int row = rb + r;
          if (row < M && col < Nn) {
            float v = acc[tm][tn][r] * alpha + bv;
            if (relu) v = fmaxf(v, 0.f);
            Db[(long)row * ldc + col] = v;
          }
        }
      }
    }
  }
}

static inline void gemm(hipStream_t st, const float* A, const float* Bm,
                        const float* bias, float* Dst, int M, int Nn, int K,
                        int lda, int ldb, int ldc, float alpha, int btrans,
                        int relu, int Z = 1, int zInner = 1, long aO = 0,
                        long aI = 0, long bO = 0, long bI = 0, long dO = 0,
                        long dI = 0) {
  dim3 g((unsigned)((Nn + 63) / 64), (unsigned)((M + 127) / 128), (unsigned)Z);
  gemm_f16_kernel<<<g, 256, 0, st>>>(A, Bm, bias, Dst, M, Nn, K, lda, ldb, ldc,
                                     alpha, btrans, relu, zInner, aO, aI, bO,
                                     bI, dO, dI);
}

// ---------------------------------------------------------------------------
// Position-embedding gather: g_c[bn, d] = pe_tab[c][clip(pos)][d]
// ---------------------------------------------------------------------------
__global__ __launch_bounds__(256)
void pe_gather_kernel(const float* __restrict__ pos, const float* __restrict__ tab,
                      float* __restrict__ g0, float* __restrict__ g1) {
  int i = blockIdx.x * 256 + threadIdx.x;
  if (i >= kB * kN * 128) return;
  int bn = i >> 7, d = i & 127;
#pragma unroll
  for (int c = 0; c < 2; ++c) {
    float v = pos[bn * 2 + c];
    v = fminf(fmaxf(v, 0.f), 1000.f);
    int id = (int)v;
    float e = tab[((long)c * 1001 + id) * 128 + d];
    if (c == 0) g0[i] = e; else g1[i] = e;
  }
}

__global__ __launch_bounds__(256)
void addhalf_kernel(const float* __restrict__ src, float* __restrict__ x, int half) {
  int i = blockIdx.x * 256 + threadIdx.x;
  if (i >= kB * kN * 128) return;
  int bn = i >> 7, d = i & 127;
  x[(long)bn * kD + half * 128 + d] += src[i];
}

// ---------------------------------------------------------------------------
// Per-batch position mean/std (population std, matches jnp std)
// ---------------------------------------------------------------------------
__global__ __launch_bounds__(256)
void posstats_kernel(const float* __restrict__ pos, float* __restrict__ pm,
                     float* __restrict__ ps) {
  __shared__ float red[256];
  int b = blockIdx.x, tid = threadIdx.x;
  for (int c = 0; c < 2; ++c) {
    float s = 0.f, s2 = 0.f;
    for (int n = tid; n < kN; n += 256) {
      float v = pos[((long)b * kN + n) * 2 + c];
      s += v; s2 += v * v;
    }
    float ts  = block_reduce_sum(s, red);
    float ts2 = block_reduce_sum(s2, red);
    if (tid == 0) {
      float mean = ts / kN;
      float var  = fmaxf(ts2 / kN - mean * mean, 0.f);
      pm[b * 2 + c] = mean;
      ps[b * 2 + c] = sqrtf(var) + 1e-8f;
    }
    __syncthreads();
  }
}

// ---------------------------------------------------------------------------
// kNN adjacency: union of top-8 (self incl.) and top-7 (self excl.)
// One 256-thread block per (b, target). Neighbor index lists -> nbr/ncnt.
// ---------------------------------------------------------------------------
__device__ __forceinline__ int block_argmin_1024(const float* dw, float* rv, int* ri) {
  int tid = threadIdx.x;
  float lm = 3e38f; int li = 0;
  for (int s = tid; s < 1024; s += 256) {
    float v = dw[s];
    if (v < lm) { lm = v; li = s; }
  }
  rv[tid] = lm; ri[tid] = li;
  __syncthreads();
  for (int off = 128; off > 0; off >>= 1) {
    if (tid < off) {
      if (rv[tid + off] < rv[tid] ||
          (rv[tid + off] == rv[tid] && ri[tid + off] < ri[tid])) {
        rv[tid] = rv[tid + off]; ri[tid] = ri[tid + off];
      }
    }
    __syncthreads();
  }
  int r = ri[0]; __syncthreads();
  return r;
}

__global__ __launch_bounds__(256)
void knn_kernel(const float* __restrict__ pos, const float* __restrict__ pm,
                const float* __restrict__ ps, int* __restrict__ nbr,
                int* __restrict__ ncnt) {
  __shared__ float dist[1024];
  __shared__ float dwork[1024];
  __shared__ float rv[256];
  __shared__ int   ri[256];
  __shared__ int   list[16];
  __shared__ int   lcnt;
  int t = blockIdx.x, b = blockIdx.y, tid = threadIdx.x;
  long bidx = (long)b * kN + t;

  float mx = pm[b * 2 + 0], my = pm[b * 2 + 1];
  float sx = ps[b * 2 + 0], sy = ps[b * 2 + 1];
  float px = (pos[bidx * 2 + 0] - mx) / sx;
  float py = (pos[bidx * 2 + 1] - my) / sy;
  for (int s = tid; s < kN; s += 256) {
    float qx = (pos[((long)b * kN + s) * 2 + 0] - mx) / sx;
    float qy = (pos[((long)b * kN + s) * 2 + 1] - my) / sy;
    float dx = px - qx, dy = py - qy;
    float d  = sqrtf(dx * dx + dy * dy + 1e-12f);
    dist[s] = d; dwork[s] = d;
  }
  if (tid == 0) lcnt = 0;
  __syncthreads();

  // local k=8 (self included: self-distance ~0 -> always picked)
  for (int it = 0; it < 8; ++it) {
    int idx = block_argmin_1024(dwork, rv, ri);
    if (tid == 0) { list[lcnt] = idx; lcnt = lcnt + 1; dwork[idx] = 3e38f; }
    __syncthreads();
  }
  // global k=7 (self excluded), union with dedupe
  for (int s = tid; s < kN; s += 256) dwork[s] = dist[s];
  __syncthreads();
  if (tid == 0) dwork[t] = 3e38f;
  __syncthreads();
  for (int it = 0; it < 7; ++it) {
    int idx = block_argmin_1024(dwork, rv, ri);
    if (tid == 0) {
      bool found = false;
      for (int j = 0; j < lcnt; ++j) if (list[j] == idx) { found = true; break; }
      if (!found) { list[lcnt] = idx; lcnt = lcnt + 1; }
      dwork[idx] = 3e38f;
    }
    __syncthreads();
  }
  if (tid == 0) ncnt[bidx] = lcnt;
  if (tid < 16) nbr[bidx * 16 + tid] = (tid < lcnt) ? list[tid] : 0;
}

// ---------------------------------------------------------------------------
// GAT: per-node attention logit halves a_src/a_dst[bn, h]
// ---------------------------------------------------------------------------
__global__ __launch_bounds__(256)
void gat_att_kernel(const float* __restrict__ hbuf, const float* __restrict__ asW,
                    const float* __restrict__ adW, float* __restrict__ asrc,
                    float* __restrict__ adst) {
  int i = blockIdx.x * 256 + threadIdx.x;
  if (i >= kB * kN * kH) return;
  int h = i & 1, bn = i >> 1;
  const float* hp = hbuf + (long)bn * kD + h * kC;
  float s = 0.f, d = 0.f;
  for (int c = 0; c < kC; ++c) {
    float v = hp[c];
    s += v * asW[h * kC + c];
    d += v * adW[h * kC + c];
  }
  asrc[i] = s; adst[i] = d;
}

// GAT aggregate + bias + residual + LayerNorm. Block = 256 threads per (b,t).
__global__ __launch_bounds__(256)
void gat_agg_kernel(const float* __restrict__ x, const float* __restrict__ hbuf,
                    const float* __restrict__ asrc, const float* __restrict__ adst,
                    const int* __restrict__ nbr, const int* __restrict__ ncnt,
                    const float* __restrict__ gb, const float* __restrict__ lng,
                    const float* __restrict__ lnb, float* __restrict__ xout) {
  __shared__ int   sidx[16];
  __shared__ float w[2][16];
  __shared__ int   scnt;
  __shared__ float red[256];
  long bt = blockIdx.x;
  int tid = threadIdx.x;
  int b = (int)(bt >> 10);

  if (tid == 0) scnt = ncnt[bt];
  if (tid < 16) sidx[tid] = nbr[bt * 16 + tid];
  __syncthreads();
  int cnt = scnt;

  if (tid < 32) {
    int hh = tid >> 4, i = tid & 15;
    if (i < cnt) {
      float e = adst[bt * kH + hh] + asrc[((long)b * kN + sidx[i]) * kH + hh];
      w[hh][i] = (e > 0.f) ? e : 0.2f * e;   // leaky_relu(0.2)
    }
  }
  __syncthreads();
  if (tid < 2) {
    float mxv = -3e38f;
    for (int i = 0; i < cnt; ++i) mxv = fmaxf(mxv, w[tid][i]);
    float sm = 0.f;
    for (int i = 0; i < cnt; ++i) { float e = expf(w[tid][i] - mxv); w[tid][i] = e; sm += e; }
    float inv = 1.f / sm;
    for (int i = 0; i < cnt; ++i) w[tid][i] *= inv;
  }
  __syncthreads();

  int hh = tid >> 7, c = tid & 127;
  float acc = 0.f;
  for (int i = 0; i < cnt; ++i)
    acc += w[hh][i] * hbuf[((long)b * kN + sidx[i]) * kD + hh * kC + c];

  float val = acc + gb[tid] + x[bt * kD + tid];
  float mean = block_reduce_sum(val, red) * (1.f / kD);
  float dv = val - mean;
  float var = block_reduce_sum(dv * dv, red) * (1.f / kD);
  xout[bt * kD + tid] = dv * rsqrtf(var + 1e-5f) * lng[tid] + lnb[tid];
}

// ---------------------------------------------------------------------------
// out[row] = LN( a[row] + res[row] ) (res nullable); arbitrary D, in-place OK
// ---------------------------------------------------------------------------
__global__ __launch_bounds__(256)
void add_ln_kernel(const float* __restrict__ a, const float* __restrict__ res,
                   const float* __restrict__ g, const float* __restrict__ be,
                   float* __restrict__ out, int Dn) {
  __shared__ float red[256];
  long row = blockIdx.x;
  int tid = threadIdx.x;
  const float* ap = a + row * Dn;
  const float* rp = res ? res + row * Dn : nullptr;
  float s = 0.f;
  for (int d = tid; d < Dn; d += 256) s += ap[d] + (rp ? rp[d] : 0.f);
  float mean = block_reduce_sum(s, red) / Dn;
  float s2 = 0.f;
  for (int d = tid; d < Dn; d += 256) {
    float v = ap[d] + (rp ? rp[d] : 0.f) - mean;
    s2 += v * v;
  }
  float var = block_reduce_sum(s2, red) / Dn;
  float inv = rsqrtf(var + 1e-5f);
  for (int d = tid; d < Dn; d += 256) {
    float v = ap[d] + (rp ? rp[d] : 0.f);
    out[row * Dn + d] = (v - mean) * inv * g[d] + be[d];
  }
}

// in-place row softmax, length L
__global__ __launch_bounds__(256)
void softmax_kernel(float* __restrict__ p, int L) {
  __shared__ float red[256];
  long row = blockIdx.x;
  float* pp = p + row * (long)L;
  int tid = threadIdx.x;
  float m = -3e38f;
  for (int i = tid; i < L; i += 256) m = fmaxf(m, pp[i]);
  m = block_reduce_max(m, red);
  float s = 0.f;
  for (int i = tid; i < L; i += 256) { float e = expf(pp[i] - m); pp[i] = e; s += e; }
  s = block_reduce_sum(s, red);
  float inv = 1.f / s;
  for (int i = tid; i < L; i += 256) pp[i] *= inv;
}

// ---------------------------------------------------------------------------
// pooled_avg[b,d] = mean_n x[b,n,d]
// ---------------------------------------------------------------------------
__global__ __launch_bounds__(256)
void avgpool_kernel(const float* __restrict__ x, float* __restrict__ out) {
  int i = blockIdx.x * 256 + threadIdx.x;
  if (i >= kB * kD) return;
  int b = i >> 8, d = i & 255;
  float s = 0.f;
  for (int n = 0; n < kN; ++n) s += x[((long)b * kN + n) * kD + d];
  out[i] = s * (1.f / kN);
}

// ---------------------------------------------------------------------------
// Sparse-mask attention pooling (NQ=4 learned queries). Block per (q, b).
// ---------------------------------------------------------------------------
__global__ __launch_bounds__(256)
void poolattn_kernel(const float* __restrict__ qg, const float* __restrict__ kbuf,
                     const float* __restrict__ vbuf, const float* __restrict__ srand,
                     float* __restrict__ outp) {
  __shared__ float lg[1024];
  __shared__ float qh[128];
  __shared__ float red[256];
  int q = blockIdx.x, b = blockIdx.y, tid = threadIdx.x;
  for (int h = 0; h < kH; ++h) {
    if (tid < 128) qh[tid] = qg[q * kD + h * kC + tid];
    __syncthreads();
    for (int n = tid; n < kN; n += 256) {
      const float* kp = kbuf + ((long)b * kN + n) * kD + h * kC;
      float dot = 0.f;
      for (int c = 0; c < kC; ++c) dot += qh[c] * kp[c];
      float mask = (srand[q * kN + n] < 0.3f) ? 0.f : -1e9f;
      lg[n] = dot * kScale + mask;
    }
    __syncthreads();
    float m = -3e38f;
    for (int n = tid; n < kN; n += 256) m = fmaxf(m, lg[n]);
    m = block_reduce_max(m, red);
    float s = 0.f;
    for (int n = tid; n < kN; n += 256) { float e = expf(lg[n] - m); lg[n] = e; s += e; }
    s = block_reduce_sum(s, red);
    float inv = 1.f / s;
    for (int n = tid; n < kN; n += 256) lg[n] *= inv;
    __syncthreads();
    if (tid < 128) {
      float acc = 0.f;
      for (int n = 0; n < kN; ++n)
        acc += lg[n] * vbuf[((long)b * kN + n) * kD + h * kC + tid];
      outp[((long)b * kNQ + q) * kD + h * kC + tid] = acc;
    }
    __syncthreads();
  }
}

__global__ __launch_bounds__(256)
void concat_kernel(const float* __restrict__ avg, const float* __restrict__ att,
                   float* __restrict__ comb) {
  int i = blockIdx.x * 256 + threadIdx.x;
  if (i >= kB * (1 + kNQ) * kD) return;
  int b = i / ((1 + kNQ) * kD), d = i % ((1 + kNQ) * kD);
  comb[i] = (d < kD) ? avg[b * kD + d] : att[b * kNQ * kD + (d - kD)];
}

// ---------------------------------------------------------------------------
// Workspace layout (floats)
// ---------------------------------------------------------------------------
constexpr long OFF_X   = 0;                       // [B,N,D]   1,048,576
constexpr long OFF_A   = 1048576;                 // big buf   4,194,304
constexpr long OFF_B   = OFF_A + 4194304;         // [B,N,D]
constexpr long OFF_C   = OFF_B + 1048576;         // [B,N,D]
constexpr long OFF_LG  = OFF_C + 1048576;         // [B,H,N,N] 8,388,608
constexpr long OFF_S   = OFF_LG + 8388608;
constexpr long S_PM    = OFF_S;
constexpr long S_PS    = S_PM + 8;
constexpr long S_ASRC  = S_PS + 8;
constexpr long S_ADST  = S_ASRC + (long)kB * kN * kH;
constexpr long S_QG    = S_ADST + (long)kB * kN * kH;
constexpr long S_PAVG  = S_QG + kNQ * kD;
constexpr long S_PATTP = S_PAVG + kB * kD;
constexpr long S_PATT  = S_PATTP + kB * kNQ * kD;
constexpr long S_COMB  = S_PATT + kB * kNQ * kD;
constexpr long S_H1    = S_COMB + kB * (1 + kNQ) * kD;
constexpr long S_H2    = S_H1 + kB * 1024;
constexpr long S_H3    = S_H2 + kB * 1024;
constexpr long S_H4    = S_H3 + kB * 1024;
constexpr long S_H5    = S_H4 + kB * 1032;
constexpr long S_H6    = S_H5 + kB * 512;
constexpr long S_NBR   = S_H6 + kB * 256;         // ints from here
constexpr long S_NCNT  = S_NBR + (long)kB * kN * 16;

// ---------------------------------------------------------------------------
extern "C" void kernel_launch(void* const* d_in, const int* in_sizes, int n_in,
                              void* d_out, int out_size, void* d_ws,
                              size_t ws_size, hipStream_t stream) {
  (void)in_sizes; (void)n_in; (void)out_size; (void)ws_size;
  const float* inputs  = (const float*)d_in[0];
  const float* pos     = (const float*)d_in[1];
  const float* srand_  = (const float*)d_in[2];
  const float* emb_W   = (const float*)d_in[3];
  const float* emb_b   = (const float*)d_in[4];
  const float* pe_tab  = (const float*)d_in[5];
  const float* pe_W1   = (const float*)d_in[6];
  const float* pe_b1   = (const float*)d_in[7];
  const float* pe_W2   = (const float*)d_in[8];
  const float* pe_b2   = (const float*)d_in[9];
  const float* gat_W   = (const float*)d_in[10];
  const float* gat_as  = (const float*)d_in[11];
  const float* gat_ad  = (const float*)d_in[12];
  const float* gat_b   = (const float*)d_in[13];
  const float* ln_g    = (const float*)d_in[14];
  const float* ln_b    = (const float*)d_in[15];
  const float* qkv_W   = (const float*)d_in[16];
  const float* qkv_b   = (const float*)d_in[17];
  const float* out_W   = (const float*)d_in[18];
  const float* out_b   = (const float*)d_in[19];
  const float* ff1_W   = (const float*)d_in[20];
  const float* ff1_b   = (const float*)d_in[21];
  const float* ff2_W   = (const float*)d_in[22];
  const float* ff2_b   = (const float*)d_in[23];
  const float* ln1_g   = (const float*)d_in[24];
  const float* ln1_b   = (const float*)d_in[25];
  const float* ln2_g   = (const float*)d_in[26];
  const float* ln2_b   = (const float*)d_in[27];
  const float* query_e = (const float*)d_in[28];
  const float* pin_W   = (const float*)d_in[29];
  const float* pin_b   = (const float*)d_in[30];
  const float* pout_W  = (const float*)d_in[31];
  const float* pout_b  = (const float*)d_in[32];
  const float* m1_W    = (const float*)d_in[33];
  const float* m1_b    = (const float*)d_in[34];
  const float* mln1g   = (const float*)d_in[35];
  const float* mln1b   = (const float*)d_in[36];
  const float* m2_W    = (const float*)d_in[37];
  const float* m2_b    = (const float*)d_in[38];
  const float* mln2g   = (const float*)d_in[39];
  const float* mln2b   = (const float*)d_in[40];
  const float* m3_W    = (const float*)d_in[41];
  const float* m3_b    = (const float*)d_in[42];
  const float* c1_W    = (const float*)d_in[43];
  const float* c1_b    = (const float*)d_in[44];
  const float* c2_W    = (const float*)d_in[45];
  const float* c2_b    = (const float*)d_in[46];
  const float* c3_W    = (const float*)d_in[47];
  const float* c3_b    = (const float*)d_in[48];
  const float* c4_W    = (const float*)d_in[49];
  const float* c4_b    = (const float*)d_in[50];

  float* out = (float*)d_out;
  float* ws  = (float*)d_ws;
  float* x    = ws + OFF_X;
  float* bufA = ws + OFF_A;
  float* bufB = ws + OFF_B;
  float* bufC = ws + OFF_C;
  float* lg   = ws + OFF_LG;
  float* pm   = ws + S_PM;
  float* ps   = ws + S_PS;
  float* asrc = ws + S_ASRC;
  float* adst = ws + S_ADST;
  float* qg   = ws + S_QG;
  float* pavg = ws + S_PAVG;
  float* pattp= ws + S_PATTP;
  float* patt = ws + S_PATT;
  float* comb = ws + S_COMB;
  float* h1   = ws + S_H1;
  float* h2   = ws + S_H2;
  float* h3   = ws + S_H3;
  float* h4   = ws + S_H4;
  float* h5   = ws + S_H5;
  float* h6   = ws + S_H6;
  int*   nbr  = (int*)(ws + S_NBR);
  int*   ncnt = (int*)(ws + S_NCNT);

  const int BN = kB * kN;   // 4096
  hipStream_t st = stream;

  // 1) token embedding: x = inputs @ emb_W^T + emb_b
  gemm(st, inputs, emb_W, emb_b, x, BN, kD, kE, kE, kE, kD, 1.f, 1, 0);

  // 2) 2D position embedding: gather -> MLP(128->128 relu ->128) -> concat add
  pe_gather_kernel<<<(BN * 128 + 255) / 256, 256, 0, st>>>(pos, pe_tab, bufA,
                                                           bufA + (long)BN * 128);
  for (int c = 0; c < 2; ++c) {
    gemm(st, bufA + (long)c * BN * 128, pe_W1 + c * 16384, pe_b1 + c * 128,
         bufB, BN, 128, 128, 128, 128, 128, 1.f, 1, 1);
    gemm(st, bufB, pe_W2 + c * 16384, pe_b2 + c * 128, bufC, BN, 128, 128, 128,
         128, 128, 1.f, 1, 0);
    addhalf_kernel<<<(BN * 128 + 255) / 256, 256, 0, st>>>(bufC, x, c);
  }

  // 3) kNN adjacency (union of local top-8 incl. self + global top-7 excl.)
  posstats_kernel<<<kB, 256, 0, st>>>(pos, pm, ps);
  knn_kernel<<<dim3(kN, kB), 256, 0, st>>>(pos, pm, ps, nbr, ncnt);

  // 4) GAT layers (ping-pong x <-> bufB)
  float* xc = x;
  float* xn = bufB;
  for (int l = 0; l < 2; ++l) {
    gemm(st, xc, gat_W + l * 65536, nullptr, bufA, BN, kD, kD, kD, kD, kD, 1.f, 1, 0);
    gat_att_kernel<<<(BN * kH + 255) / 256, 256, 0, st>>>(
        bufA, gat_as + l * kH * kC, gat_ad + l * kH * kC, asrc, adst);
    gat_agg_kernel<<<BN, 256, 0, st>>>(xc, bufA, asrc, adst, nbr, ncnt,
                                       gat_b + l * kD, ln_g, ln_b, xn);
    float* t = xc; xc = xn; xn = t;
  }
  // after 2 swaps, xc == x again

  // 5) Transformer encoder layers
  for (int l = 0; l < 2; ++l) {
    // qkv: [BN, 768]
    gemm(st, x, qkv_W + (long)l * 768 * kD, qkv_b + l * 768, bufA, BN, 3 * kD,
         kD, kD, kD, 3 * kD, 1.f, 1, 0);
    // logits[b,h] = Q @ K^T * scale  (batched over z = b*H + h)
    gemm(st, bufA, bufA + 256, nullptr, lg, kN, kN, kC, 768, 768, kN, kScale, 1,
         0, kB * kH, kH, (long)kN * 768, 128, (long)kN * 768, 128,
         2L * kN * kN, (long)kN * kN);
    softmax_kernel<<<kB * kH * kN, 256, 0, st>>>(lg, kN);
    // attn = P @ V  -> bufC[b, n, h*C + c]
    gemm(st, lg, bufA + 512, nullptr, bufC, kN, kC, kN, kN, 768, kD, 1.f, 0, 0,
         kB * kH, kH, 2L * kN * kN, (long)kN * kN, (long)kN * 768, 128,
         (long)kN * kD, 128);
    // out proj
    gemm(st, bufC, out_W + l * 65536, out_b + l * kD, bufB, BN, kD, kD, kD, kD,
         kD, 1.f, 1, 0);
    add_ln_kernel<<<BN, 256, 0, st>>>(x, bufB, ln1_g + l * kD, ln1_b + l * kD, x, kD);
    // FFN
    gemm(st, x, ff1_W + (long)l * kFF * kD, ff1_b + l * kFF, bufA, BN, kFF, kD,
         kD, kD, kFF, 1.f, 1, 1);
    gemm(st, bufA, ff2_W + (long)l * kD * kFF, ff2_b + l * kD, bufB, BN, kD,
         kFF, kFF, kFF, kD, 1.f, 1, 0);
    add_ln_kernel<<<BN, 256, 0, st>>>(x, bufB, ln2_g + l * kD, ln2_b + l * kD, x, kD);
  }

  // 6) final LN + average pool
  add_ln_kernel<<<BN, 256, 0, st>>>(x, nullptr, ln_g, ln_b, x, kD);
  avgpool_kernel<<<(kB * kD + 255) / 256, 256, 0, st>>>(x, pavg);

  // 7) attention pooling with learned queries
  gemm(st, query_e, pin_W, pin_b, qg, kNQ, kD, kD, kD, kD, kD, 1.f, 1, 0);
  gemm(st, x, pin_W + 65536, pin_b + kD, bufB, BN, kD, kD, kD, kD, kD, 1.f, 1, 0);
  gemm(st, x, pin_W + 131072, pin_b + 2 * kD, bufC, BN, kD, kD, kD, kD, kD, 1.f, 1, 0);
  poolattn_kernel<<<dim3(kNQ, kB), 256, 0, st>>>(qg, bufB, bufC, srand_, pattp);
  gemm(st, pattp, pout_W, pout_b, patt, kB * kNQ, kD, kD, kD, kD, kD, 1.f, 1, 0);
  concat_kernel<<<(kB * (1 + kNQ) * kD + 255) / 256, 256, 0, st>>>(pavg, patt, comb);

  // 8) MLP head + classifier
  gemm(st, comb, m1_W, m1_b, h1, kB, 1024, (1 + kNQ) * kD, (1 + kNQ) * kD,
       (1 + kNQ) * kD, 1024, 1.f, 1, 1);
  add_ln_kernel<<<kB, 256, 0, st>>>(h1, nullptr, mln1g, mln1b, h1, 1024);
  gemm(st, h1, m2_W, m2_b, h2, kB, 1024, 1024, 1024, 1024, 1024, 1.f, 1, 1);
  add_ln_kernel<<<kB, 256, 0, st>>>(h2, nullptr, mln2g, mln2b, h2, 1024);
  gemm(st, h2, m3_W, m3_b, h3, kB, 1024, 1024, 1024, 1024, 1024, 1.f, 1, 0);
  gemm(st, h3, c1_W, c1_b, h4, kB, 1032, 1024, 1024, 1024, 1032, 1.f, 1, 1);
  gemm(st, h4, c2_W, c2_b, h5, kB, 512, 1032, 1032, 1032, 512, 1.f, 1, 1);
  gemm(st, h5, c3_W, c3_b, h6, kB, 256, 512, 512, 512, 256, 1.f, 1, 1);
  gemm(st, h6, c4_W, c4_b, out, kB, kNCLS, 256, 256, 256, kNCLS, 1.f, 1, 0);
}
